// FourierLayer_45346264711543
// MI455X (gfx1250) — compile-verified, matrix operand order checked by reading, not run on hardware
//
#include <hip/hip_runtime.h>
#include <hip/hip_bf16.h>
#include <cstddef>

typedef __attribute__((ext_vector_type(16))) _Float16 v16h;
typedef __attribute__((ext_vector_type(4)))  _Float16 v4h;
typedef __attribute__((ext_vector_type(8)))  float    v8f;

constexpr int B_ = 16, S_ = 8192, H_ = 256, MODES_ = 16;
constexpr float PI2 = 6.28318530717958647692f;

// K-index of half-element i (0..15) of a v16h fragment, given lane-half
// (lane>>4). CDNA5 ISA 7.12.2, 16-bit A 16x32 (B 32x16 mirrored):
//   i<8 : K = i      + 8*half
//   i>=8: K = 16+i-8 + 8*half
__device__ __forceinline__ int kmap(int i, int half) {
    return (i < 8) ? (i + 8 * half) : (16 + (i - 8) + 8 * half);
}
// Inverse: K j (0..31) -> (half, i)
__device__ __forceinline__ void inv_kmap(int j, int* half, int* i) {
    int h  = (j >> 3) & 1;
    int j2 = j - 8 * h;                 // in {0..7} U {16..23}
    *half = h;
    *i    = (j2 < 8) ? j2 : (j2 - 8);
}

__device__ __forceinline__ v8f wmma_f16(v16h a, v16h b, v8f c) {
    return __builtin_amdgcn_wmma_f32_16x16x32_f16(false, a, false, b, (short)0, c,
                                                  false, false);
}

// ---------------------------------------------------------------------------
// Kernel 0: truncated-DFT basis in f16, stored directly in A-FRAGMENT ORDER:
//   F[sblk][half][m][i]  with  s = sblk*32 + kmap(i, half)
// so kernel A loads each A fragment as one aligned 32-byte v16h load.
// Fre = cos(2pi m s/S), Fim = -sin(2pi m s/S)  (forward rfft sign).
// ---------------------------------------------------------------------------
__global__ void __launch_bounds__(256)
basis_kernel(_Float16* __restrict__ FreF, _Float16* __restrict__ FimF) {
    int idx = blockIdx.x * blockDim.x + threadIdx.x;
    const int total = MODES_ * S_;                   // 131072 per array
    for (; idx < total; idx += gridDim.x * blockDim.x) {
        int nblk = idx >> 9;
        int rem  = idx & 511;
        int half = rem >> 8;
        int m    = (rem >> 4) & 15;
        int i    = rem & 15;
        int s    = nblk * 32 + kmap(i, half);
        float ang = PI2 * (float)m * (float)s / (float)S_;
        float sn, cs;
        __sincosf(ang, &sn, &cs);
        FreF[idx] = (_Float16)cs;
        FimF[idx] = (_Float16)(-sn);
    }
}

// ---------------------------------------------------------------------------
// Kernel A: forward truncated DFT.  X{re,im}[b][m][h] += sum_s F[m][s]*x[b][s][h]
// Grid (b=16, hblk=4, schunk=8), block 128 = 4 waves (one 16-wide h tile each).
// x tile (32 s x 64 h) is staged f32->f16 into LDS **in B-fragment order**
// [half][n][i], double-buffered; B fragment read = one 32B v16h (2x b128).
// A fragments come straight from the fragment-ordered basis (2x b128).
// Partial sums across schunks combine via global_atomic_add_f32.
// ---------------------------------------------------------------------------
__device__ __forceinline__ void stage_tile(const float* __restrict__ xbase,
                                           _Float16 (*__restrict__ buf)[64][16],
                                           int t) {
    const int n  = t & 63;              // h within 64-wide block
    const int qs = t >> 6;              // 0..1
#pragma unroll
    for (int r = 0; r < 4; ++r) {
        int q = qs * 4 + r;             // K-quad 0..7  (K = 4q..4q+3)
        const float* p = xbase + (size_t)(4 * q) * H_ + n;  // lane-coalesced
        float f0 = p[0];
        float f1 = p[H_];
        float f2 = p[2 * H_];
        float f3 = p[3 * H_];
        v4h h;
        h[0] = (_Float16)f0; h[1] = (_Float16)f1;
        h[2] = (_Float16)f2; h[3] = (_Float16)f3;
        int hq = (q >> 1) & 1;                      // fragment half
        int i0 = 4 * (q & 1) + 8 * (q >> 2);        // consecutive i positions
        *(v4h*)&buf[hq][n][i0] = h;                 // ds_store_b64
    }
}

__global__ void __launch_bounds__(128)
dft_fwd_kernel(const float* __restrict__ x, const _Float16* __restrict__ FreF,
               const _Float16* __restrict__ FimF, float* __restrict__ Xre,
               float* __restrict__ Xim) {
    __shared__ alignas(32) _Float16 shb[2][2][64][16];   // double buffer, 8 KiB

    const int b      = blockIdx.x;
    const int hblk   = blockIdx.y;
    const int schunk = blockIdx.z;
    const int t      = threadIdx.x;
    const int lane   = t & 31;
    const int wave   = t >> 5;
    const int half   = (lane >> 4) & 1;
    const int mn     = lane & 15;       // A row (mode) / B col (h)

    const float* xblk = x + (size_t)b * S_ * H_ + hblk * 64;
    const int sBeg = schunk * 1024, sEnd = sBeg + 1024;

    v8f accre = {}, accim = {};

    stage_tile(xblk + (size_t)sBeg * H_, shb[0], t);
    int cur = 0;
#pragma unroll 1
    for (int s0 = sBeg; s0 < sEnd; s0 += 32, cur ^= 1) {
        __syncthreads();                              // buf[cur] ready
        if (s0 + 32 < sEnd) {
            stage_tile(xblk + (size_t)(s0 + 32) * H_, shb[cur ^ 1], t);
            __builtin_prefetch(xblk + (size_t)(s0 + 64 + (t >> 6)) * H_ + (t & 63),
                               0, 0);                 // global_prefetch_b8
        }
        size_t foff = ((size_t)(s0 >> 5) << 9) + (half << 8) + (mn << 4);
        v16h are = *(const v16h*)(FreF + foff);       // 2x global_load_b128
        v16h aim = *(const v16h*)(FimF + foff);
        v16h bx  = *(const v16h*)&shb[cur][half][wave * 16 + mn][0];  // 2x ds_load_b128
        accre = wmma_f16(are, bx, accre);
        accim = wmma_f16(aim, bx, accim);
    }

    const int hout = hblk * 64 + wave * 16 + mn;
#pragma unroll
    for (int r = 0; r < 8; ++r) {
        int m = r + 8 * half;                         // C/D: M = vgpr + 8*lane[4]
        atomicAdd(&Xre[(size_t)b * MODES_ * H_ + (size_t)m * H_ + hout], accre[r]);
        atomicAdd(&Xim[(size_t)b * MODES_ * H_ + (size_t)m * H_ + hout], accim[r]);
    }
}

// ---------------------------------------------------------------------------
// Kernel B: per-mode complex mix  Y[b,k] = sum_h X[b,h]*W[h,k]  (M = batch = 16).
// Yre += Are*Bre + (-Aim)*Bim ; Yim += Are*Bim + Aim*Bre  (f16 WMMA has no A/B
// NEG, so -Aim is materialized). Results are stored as f16 DIRECTLY in the
// inverse-DFT B-fragment layout  Yfrag[b][k][half][i]:
//   basis index j = mode       -> (half,i) holds Yre
//   basis index j = 16 + mode  -> (half,i) holds Yim
// Grid (mode=16, ktile=16), one wave per block, 8 K-steps over H=256.
// ---------------------------------------------------------------------------
__global__ void __launch_bounds__(32)
modemix_kernel(const float* __restrict__ wr, const float* __restrict__ wi,
               const float* __restrict__ Xre, const float* __restrict__ Xim,
               _Float16* __restrict__ Yfrag) {
    const int mode = blockIdx.x;
    const int k0   = blockIdx.y * 16;
    const int lane = threadIdx.x & 31;
    const int half = lane >> 4;
    const int mn   = lane & 15;         // A row (batch) / B col (k)

    v8f accre = {}, accim = {};
#pragma unroll 1
    for (int h0 = 0; h0 < H_; h0 += 32) {
        float xr[16], xi[16], wrv[16], wiv[16];
#pragma unroll
        for (int i = 0; i < 16; ++i) {
            int h = h0 + kmap(i, half);
            xr[i]  = Xre[(size_t)mn * MODES_ * H_ + (size_t)mode * H_ + h];
            xi[i]  = Xim[(size_t)mn * MODES_ * H_ + (size_t)mode * H_ + h];
            wrv[i] = wr[((size_t)h * H_ + (k0 + mn)) * MODES_ + mode];
            wiv[i] = wi[((size_t)h * H_ + (k0 + mn)) * MODES_ + mode];
        }
        v16h ar, ai, ain, br, bi;
#pragma unroll
        for (int i = 0; i < 16; ++i) {
            ar[i]  = (_Float16)xr[i];
            ai[i]  = (_Float16)xi[i];
            ain[i] = (_Float16)(-xi[i]);
            br[i]  = (_Float16)wrv[i];
            bi[i]  = (_Float16)wiv[i];
        }
        accre = wmma_f16(ar,  br, accre);
        accre = wmma_f16(ain, bi, accre);
        accim = wmma_f16(ar,  bi, accim);
        accim = wmma_f16(ai,  br, accim);
    }

    int hre, ire, him, iim;
    inv_kmap(mode,      &hre, &ire);
    inv_kmap(16 + mode, &him, &iim);
#pragma unroll
    for (int r = 0; r < 8; ++r) {
        int brow = r + 8 * half;        // batch index
        size_t base = ((size_t)brow * H_ + (k0 + mn)) * 32;
        Yfrag[base + hre * 16 + ire] = (_Float16)accre[r];
        Yfrag[base + him * 16 + iim] = (_Float16)accim[r];
    }
}

// ---------------------------------------------------------------------------
// Kernel C: inverse truncated DFT + exact GELU.
//   y[b,s,k] = (1/S) * sum_{j=0..31} basis[s,j] * Yj[b,k]
//   j<16 : basis = (j==0?1:2)*cos(2pi j s/S),  Yj = Yre[j]
//   j>=16: basis = (m==0?0:-2*sin(2pi m s/S)), Yj = Yim[m], m=j-16
// K=32 => ONE WMMA per 16x16 output tile. Basis built in LDS in A-fragment
// order (one v16h read); Y fragments are single 32B loads from Yfrag.
// Grid (stile=512, b=16), block 128 = 4 waves x 4 k-tiles.
// ---------------------------------------------------------------------------
__global__ void __launch_bounds__(128)
idft_gelu_kernel(const _Float16* __restrict__ Yfrag, float* __restrict__ out) {
    __shared__ alignas(32) _Float16 bas[2][16][16];  // [half][s_local][i]

    const int s0 = blockIdx.x * 16;
    const int b  = blockIdx.y;
    const int t  = threadIdx.x;

#pragma unroll
    for (int r = 0; r < 4; ++r) {
        int idx  = t + r * 128;                      // 0..511
        int half = idx >> 8;
        int m    = (idx >> 4) & 15;                  // s row
        int i    = idx & 15;
        int j    = kmap(i, half);
        int md   = j & 15;
        float ang = PI2 * (float)md * (float)(s0 + m) / (float)S_;
        float sn, cs;
        __sincosf(ang, &sn, &cs);
        float val = (j < 16) ? ((md == 0) ? cs : 2.0f * cs)
                             : ((md == 0) ? 0.0f : -2.0f * sn);
        bas[half][m][i] = (_Float16)val;
    }
    __syncthreads();

    const int lane = t & 31, wave = t >> 5;
    const int half = lane >> 4, mn = lane & 15;

    v16h afrag = *(const v16h*)&bas[half][mn][0];    // 2x ds_load_b128
    const float inv = 1.0f / (float)S_;

#pragma unroll 1
    for (int kt = 0; kt < 4; ++kt) {
        int k0 = (wave * 4 + kt) * 16;
        v16h bfrag = *(const v16h*)
            &Yfrag[((size_t)b * H_ + (k0 + mn)) * 32 + half * 16];  // 2x b128
        v8f acc = {};
        acc = wmma_f16(afrag, bfrag, acc);
#pragma unroll
        for (int r = 0; r < 8; ++r) {
            int sl = r + 8 * half;
            float v = acc[r] * inv;
            float g = 0.5f * v * (1.0f + erff(v * 0.70710678118654752f));
            out[(size_t)b * S_ * H_ + (size_t)(s0 + sl) * H_ + k0 + mn] = g;
        }
    }
}

// ---------------------------------------------------------------------------
extern "C" void kernel_launch(void* const* d_in, const int* in_sizes, int n_in,
                              void* d_out, int out_size, void* d_ws, size_t ws_size,
                              hipStream_t stream) {
    const float* x  = (const float*)d_in[0];  // [B,S,H] f32
    const float* wr = (const float*)d_in[1];  // [H,H,MODES] f32
    const float* wi = (const float*)d_in[2];  // [H,H,MODES] f32
    float* out = (float*)d_out;               // [B,S,H] f32
    (void)in_sizes; (void)n_in; (void)out_size; (void)ws_size;

    // Workspace (1.25 MiB total, fully L2-resident):
    //   [0,    256K) FreF  f16, A-fragment order [256][2][16][16]
    //   [256K, 512K) FimF  f16, same layout
    //   [512K, 768K) Xre   f32[16][16][256]
    //   [768K, 1M  ) Xim   f32[16][16][256]
    //   [1M,   1.25M) Yfrag f16[16][256][2][16]  (iDFT B-fragment order)
    char* ws = (char*)d_ws;
    _Float16* FreF  = (_Float16*)(ws);
    _Float16* FimF  = (_Float16*)(ws + 256 * 1024);
    float*    Xre   = (float*)   (ws + 512 * 1024);
    float*    Xim   = (float*)   (ws + 768 * 1024);
    _Float16* Yfrag = (_Float16*)(ws + 1024 * 1024);

    hipMemsetAsync(Xre, 0, 2 * 256 * 1024, stream);  // zero atomic accumulators

    basis_kernel<<<256, 256, 0, stream>>>(FreF, FimF);
    dft_fwd_kernel<<<dim3(16, 4, 8), 128, 0, stream>>>(x, FreF, FimF, Xre, Xim);
    modemix_kernel<<<dim3(16, 16), 32, 0, stream>>>(wr, wi, Xre, Xim, Yfrag);
    idft_gelu_kernel<<<dim3(512, 16), 128, 0, stream>>>(Yfrag, out);
}
